// Ch_Pos_GCL_62122406969661
// MI455X (gfx1250) — matrix operand order repaired and checked by reading, not compiled, not run on hardware
//
#include <hip/hip_runtime.h>
#include <hip/hip_bf16.h>

// ---------------------------------------------------------------------------
// MI455X (gfx1250) fused edge-MLP for the Ch_Pos_GCL layer.
//
// Roofline: ~65 GFLOP of 128-wide GEMMs vs ~0.9 GB of HBM traffic
// (gathered h + three [E,128] outputs). At 23.3 TB/s the memory floor is
// ~40us, so the matmuls must run on the bf16 WMMA pipe (16x16x32, f32 acc).
// All weights (200 KB bf16) live in LDS (CDNA5: 320 KB/WGP), loaded once per
// workgroup and reused across ~39 16-edge tiles (20000 tiles / 512 WGs).
//
// Round-1 fix: weights are stored in *fragment-major* order so every WMMA
// B-fragment is 32 contiguous LDS bytes (2x ds_load_b128) instead of 16
// strided ds_load_u16 + 16 v_mov_b16 packing ops.
// ---------------------------------------------------------------------------

#define NN 20000
#define NE 320000
#define HID 128
#define EPSF 1e-8f

typedef __attribute__((ext_vector_type(16))) __bf16 v16bf;
typedef __attribute__((ext_vector_type(8)))  __bf16 v8bf;
typedef __attribute__((ext_vector_type(8)))  float  v8f;

__device__ __forceinline__ unsigned short f2bf(float f) {
  unsigned int u = __builtin_bit_cast(unsigned int, f);
  u += 0x7fffu + ((u >> 16) & 1u);           // round-to-nearest-even
  return (unsigned short)(u >> 16);
}
__device__ __forceinline__ float silu_f(float x) {
  return x * (1.0f / (1.0f + __expf(-x)));
}

// A-matrix fragment: 16x32 bf16, lane L supplies row M=L%16.
// v16bf element e <-> K = k0 + (L/16)*8 + (e<8 ? e : e+8)   (ISA 7.12.2)
// Both 8-element halves are contiguous 16-byte chunks -> 2x ds_load_b128.
__device__ __forceinline__ v16bf load_a_frag(const unsigned short* __restrict__ A,
                                             int M, int lda, int k0, int half) {
  const v8bf* p = (const v8bf*)(A + M * lda + k0 + half * 8);
  v8bf lo = p[0];   // K = k0+half*8 + 0..7
  v8bf hi = p[2];   // K = k0+half*8 + 16..23
  return __builtin_shufflevector(lo, hi, 0, 1, 2, 3, 4, 5, 6, 7,
                                 8, 9, 10, 11, 12, 13, 14, 15);
}

// B-matrix fragment from fragment-major packed weights: lane's 16 bf16 values
// are contiguous (32B aligned) -> 2x ds_load_b128, no VALU packing.
// Packed layout: element ((nt*KT + kt)*32 + lane)*16 + e  where
//   K = kt*32 + (lane/16)*8 + (e<8 ? e : e+8),  N = nt*16 + (lane%16).
__device__ __forceinline__ v16bf load_b_packed(const unsigned short* __restrict__ Wp,
                                               int nt, int KT, int kt, int lane) {
  return *(const v16bf*)(Wp + (size_t)(((nt * KT + kt) * 32 + lane) << 4));
}

__device__ __forceinline__ v8f wmma_bf16(v16bf a, v16bf b, v8f c) {
  return __builtin_amdgcn_wmma_f32_16x16x32_bf16(false, a, false, b,
                                                 (short)0, c, false, false);
}

// ---------------------------------------------------------------------------
// Kernel 1: zero segment-sum scratch + convert weights to bf16 in
// fragment-major packed order.
// Packed bf16 segment bases (element offsets):
//   wce1 [K=256] @0, wce2 [K=128] @32768, wpe1 [K=32, rows>=18 zero] @49152,
//   wpe2 @53248, ws1 @69632, ws2 @86016  -> 102400 elements total.
// ---------------------------------------------------------------------------
__device__ __forceinline__ float pick_w(const float* __restrict__ W, int K,
                                        int j, bool pe1pad) {
  int e  = j & 15;
  int q  = j >> 4;
  int l  = q & 31;
  int f  = q >> 5;
  int KT = K >> 5;
  int nt = f / KT;
  int kt = f - nt * KT;
  int k  = kt * 32 + (l >> 4) * 8 + (e < 8 ? e : e + 8);
  int n  = nt * 16 + (l & 15);
  if (pe1pad && k >= 18) return 0.0f;
  return W[k * HID + n];
}

__global__ void k_prep(const float* __restrict__ Wce1, const float* __restrict__ Wce2,
                       const float* __restrict__ Wpe1, const float* __restrict__ Wpe2,
                       const float* __restrict__ Ws1,  const float* __restrict__ Ws2,
                       float* __restrict__ sums_cnt, unsigned short* __restrict__ wbf) {
  int i = blockIdx.x * blockDim.x + threadIdx.x;
  if (i < NN * 10) sums_cnt[i] = 0.0f;
  if (i < 102400) {
    float v;
    if (i < 32768)       v = pick_w(Wce1, 256, i, false);
    else if (i < 49152)  v = pick_w(Wce2, 128, i - 32768, false);
    else if (i < 53248)  v = pick_w(Wpe1, 32,  i - 49152, true);
    else if (i < 69632)  v = pick_w(Wpe2, 128, i - 53248, false);
    else if (i < 86016)  v = pick_w(Ws1,  128, i - 69632, false);
    else                 v = pick_w(Ws2,  128, i - 86016, false);
    wbf[i] = f2bf(v);
  }
}

// ---------------------------------------------------------------------------
// Kernel 2: per-edge geometry + f32 atomic segment sums (9 comps + count).
// ---------------------------------------------------------------------------
__global__ void k_geom(const float* __restrict__ coord, const long long* __restrict__ edges,
                       float* __restrict__ sums, float* __restrict__ cnt,
                       float* __restrict__ out_cdn) {
  int e = blockIdx.x * blockDim.x + threadIdx.x;
  if (e >= NE) return;
  int r = (int)edges[e];
  int c = (int)edges[NE + e];
  float dx = coord[3 * r + 0] - coord[3 * c + 0];
  float dy = coord[3 * r + 1] - coord[3 * c + 1];
  float dz = coord[3 * r + 2] - coord[3 * c + 2];
  float r2 = dx * dx + dy * dy + dz * dz;
  float dn = sqrtf(r2);
  float invn = 1.0f / (dn + EPSF);
  out_cdn[3 * e + 0] = dx * invn;
  out_cdn[3 * e + 1] = dy * invn;
  out_cdn[3 * e + 2] = dz * invn;
  float idn = 1.0f / dn;                 // no self-loops guaranteed
  float i3 = idn * idn * idn;
  float i4 = i3 * idn;
  float i5 = i4 * idn;
  atomicAdd(&sums[9 * r + 0], i3 * dx);
  atomicAdd(&sums[9 * r + 1], i3 * dy);
  atomicAdd(&sums[9 * r + 2], i3 * dz);
  atomicAdd(&sums[9 * r + 3], i4 * dx);
  atomicAdd(&sums[9 * r + 4], i4 * dy);
  atomicAdd(&sums[9 * r + 5], i4 * dz);
  atomicAdd(&sums[9 * r + 6], i5 * dx);
  atomicAdd(&sums[9 * r + 7], i5 * dy);
  atomicAdd(&sums[9 * r + 8], i5 * dz);
  atomicAdd(&cnt[r], 1.0f);
}

// ---------------------------------------------------------------------------
// Kernel 3: per-node normalization of accumulated force vectors.
// ---------------------------------------------------------------------------
__global__ void k_node(const float* __restrict__ sums, const float* __restrict__ cnt,
                       float* __restrict__ nvecs) {
  int n = blockIdx.x * blockDim.x + threadIdx.x;
  if (n >= NN) return;
  float dnm = fmaxf(cnt[n], 1.0f);
#pragma unroll
  for (int a = 0; a < 3; ++a) {
    float x = sums[9 * n + 3 * a + 0] / dnm;
    float y = sums[9 * n + 3 * a + 1] / dnm;
    float z = sums[9 * n + 3 * a + 2] / dnm;
    float il = 1.0f / (sqrtf(x * x + y * y + z * z) + EPSF);
    nvecs[9 * n + 3 * a + 0] = x * il;
    nvecs[9 * n + 3 * a + 1] = y * il;
    nvecs[9 * n + 3 * a + 2] = z * il;
  }
}

// ---------------------------------------------------------------------------
// Kernel 4: fused WMMA edge-MLP chain. 256 threads = 8 waves; each wave owns
// one 16-wide column tile of the 128 outputs; tiles of 16 edges per step.
// Dynamic LDS layout (bytes):
//   0      wce1(64K) 65536 wce2(32K) 98304 wpe1(8K) 106496 wpe2(32K)
//   139264 ws1(32K)  172032 ws2(32K) 204800 biases(3K) 207872 watt(512)
//   208384 A_chem(8K bf16) 216576 A_pos(1K) 217600 midA(4K) 221696 chemB(4K)
//   225792 scratch(8K f32) 233984 attv(64) 234048 idx(128)   total 234176
// ---------------------------------------------------------------------------
#define SMEM_BYTES 234176

__global__ void __launch_bounds__(256)
k_edge_mlp(const float* __restrict__ h, const float* __restrict__ coord,
           const long long* __restrict__ edges, const float* __restrict__ nvecs,
           const unsigned short* __restrict__ wbf,
           const float* __restrict__ bce1, const float* __restrict__ bce2,
           const float* __restrict__ bpe1, const float* __restrict__ bpe2,
           const float* __restrict__ bs1,  const float* __restrict__ bs2,
           const float* __restrict__ watt, const float* __restrict__ batt,
           float* __restrict__ out0, float* __restrict__ out_chem,
           float* __restrict__ out_pos) {
  extern __shared__ char smem[];
  unsigned short* w_all = (unsigned short*)smem;
  unsigned short* wce1  = w_all;
  unsigned short* wce2  = w_all + 32768;
  unsigned short* wpe1  = w_all + 49152;
  unsigned short* wpe2  = w_all + 53248;
  unsigned short* wsm1  = w_all + 69632;
  unsigned short* wsm2  = w_all + 86016;
  float*          bias   = (float*)(smem + 204800);
  float*          l_watt = (float*)(smem + 207872);
  unsigned short* A_chem = (unsigned short*)(smem + 208384);
  unsigned short* A_pos  = (unsigned short*)(smem + 216576);
  unsigned short* midA   = (unsigned short*)(smem + 217600);
  unsigned short* chemB  = (unsigned short*)(smem + 221696);
  float*          scr    = (float*)(smem + 225792);
  float*          attv   = (float*)(smem + 233984);
  int*            idx    = (int*)(smem + 234048);

  const int tid = threadIdx.x;

  // Load all packed weights/biases into LDS once per workgroup (b64 copies).
  for (int i = tid; i < 102400 / 4; i += 256)
    ((unsigned long long*)w_all)[i] = ((const unsigned long long*)wbf)[i];
  for (int i = tid; i < HID; i += 256) {
    bias[i]           = bce1[i];
    bias[HID + i]     = bce2[i];
    bias[2 * HID + i] = bpe1[i];
    bias[3 * HID + i] = bpe2[i];
    bias[4 * HID + i] = bs1[i];
    bias[5 * HID + i] = bs2[i];
    l_watt[i]         = watt[i];
  }
  __syncthreads();

  const int wave = tid >> 5;
  const int lane = tid & 31;
  const int half = lane >> 4;
  const int lrow = lane & 15;          // A-row / C-column index within tile
  const int ncol = wave * 16 + lrow;   // output column handled by this lane
  const float battv = batt[0];
  const int ntiles = NE / 16;

  for (int tile = blockIdx.x; tile < ntiles; tile += gridDim.x) {
    const int e0 = tile * 16;

    // Phase 1: per-edge geometry features -> pos_in (bf16, padded to K=32).
    if (tid < 16) {
      int m = tid, e = e0 + m;
      int r = (int)edges[e];
      int c = (int)edges[NE + e];
      idx[2 * m] = r;
      idx[2 * m + 1] = c;
      float dx = coord[3 * r + 0] - coord[3 * c + 0];
      float dy = coord[3 * r + 1] - coord[3 * c + 1];
      float dz = coord[3 * r + 2] - coord[3 * c + 2];
      float r2 = dx * dx + dy * dy + dz * dz;
      unsigned int* ap32 = (unsigned int*)(A_pos + m * 32);
      unsigned short v18[18];
#pragma unroll
      for (int a = 0; a < 3; ++a) {
        float p = nvecs[9 * r + 3 * a + 0] * nvecs[9 * c + 3 * a + 0]
                + nvecs[9 * r + 3 * a + 1] * nvecs[9 * c + 3 * a + 1]
                + nvecs[9 * r + 3 * a + 2] * nvecs[9 * c + 3 * a + 2];
        v18[a] = f2bf(p);
      }
      float sc = -0.5f;
#pragma unroll
      for (int s = 0; s < 15; ++s) {
        v18[3 + s] = f2bf(__expf(r2 * sc));
        sc *= (1.0f / 2.25f);
      }
#pragma unroll
      for (int j = 0; j < 9; ++j)
        ap32[j] = (unsigned int)v18[2 * j] | ((unsigned int)v18[2 * j + 1] << 16);
#pragma unroll
      for (int j = 9; j < 16; ++j) ap32[j] = 0u;
    }
    __syncthreads();

    // Phase 2: gather chem_in = [h[row] | h[col]] as packed bf16 pairs.
    for (int i = tid; i < 16 * 128; i += 256) {
      int m = i >> 7, p = i & 127;                   // pair index 0..127
      int node = (p < 64) ? idx[2 * m] : idx[2 * m + 1];
      const float* hp = h + (size_t)node * HID + ((p & 63) << 1);
      float x = hp[0], y = hp[1];
      ((unsigned int*)A_chem)[i] =
          (unsigned int)f2bf(x) | ((unsigned int)f2bf(y) << 16);
    }
    __syncthreads();

    // Phase 3: chem_mid = silu(chem_in @ Wce1 + bce1)   (K=256, KT=8)
    {
      v8f acc = {0.f, 0.f, 0.f, 0.f, 0.f, 0.f, 0.f, 0.f};
#pragma unroll
      for (int kt = 0; kt < 8; ++kt)
        acc = wmma_bf16(load_a_frag(A_chem, lrow, 256, kt * 32, half),
                        load_b_packed(wce1, wave, 8, kt, lane), acc);
#pragma unroll
      for (int rI = 0; rI < 8; ++rI) {
        int m = rI + half * 8;
        midA[m * HID + ncol] = f2bf(silu_f(acc[rI] + bias[ncol]));
      }
    }
    __syncthreads();

    // Phase 4: chem = chem_mid @ Wce2 + bce2  (K=128), store f32 + bf16 copy.
    {
      v8f acc = {0.f, 0.f, 0.f, 0.f, 0.f, 0.f, 0.f, 0.f};
#pragma unroll
      for (int kt = 0; kt < 4; ++kt)
        acc = wmma_bf16(load_a_frag(midA, lrow, HID, kt * 32, half),
                        load_b_packed(wce2, wave, 4, kt, lane), acc);
#pragma unroll
      for (int rI = 0; rI < 8; ++rI) {
        int m = rI + half * 8;
        float v = acc[rI] + bias[HID + ncol];
        out_chem[(size_t)(e0 + m) * HID + ncol] = v;
        chemB[m * HID + ncol] = f2bf(v);
      }
    }
    __syncthreads();

    // Phase 5: pos_mid = silu(pos_in @ Wpe1 + bpe1)   (K=32, zero padded)
    {
      v8f acc = {0.f, 0.f, 0.f, 0.f, 0.f, 0.f, 0.f, 0.f};
      acc = wmma_bf16(load_a_frag(A_pos, lrow, 32, 0, half),
                      load_b_packed(wpe1, wave, 1, 0, lane), acc);
#pragma unroll
      for (int rI = 0; rI < 8; ++rI) {
        int m = rI + half * 8;
        midA[m * HID + ncol] = f2bf(silu_f(acc[rI] + bias[2 * HID + ncol]));
      }
    }
    __syncthreads();

    // Phase 6: pos = pos_mid @ Wpe2 + bpe2  (K=128); keep tile in registers.
    v8f posr;
    {
      v8f acc = {0.f, 0.f, 0.f, 0.f, 0.f, 0.f, 0.f, 0.f};
#pragma unroll
      for (int kt = 0; kt < 4; ++kt)
        acc = wmma_bf16(load_a_frag(midA, lrow, HID, kt * 32, half),
                        load_b_packed(wpe2, wave, 4, kt, lane), acc);
#pragma unroll
      for (int rI = 0; rI < 8; ++rI) {
        int m = rI + half * 8;
        posr[rI] = acc[rI] + bias[3 * HID + ncol];
        out_pos[(size_t)(e0 + m) * HID + ncol] = posr[rI];
      }
    }
    __syncthreads();

    // Phase 7: s_mid = silu(chem_bf16 @ Ws1 + bs1)   (K=128)
    {
      v8f acc = {0.f, 0.f, 0.f, 0.f, 0.f, 0.f, 0.f, 0.f};
#pragma unroll
      for (int kt = 0; kt < 4; ++kt)
        acc = wmma_bf16(load_a_frag(chemB, lrow, HID, kt * 32, half),
                        load_b_packed(wsm1, wave, 4, kt, lane), acc);
#pragma unroll
      for (int rI = 0; rI < 8; ++rI) {
        int m = rI + half * 8;
        midA[m * HID + ncol] = f2bf(silu_f(acc[rI] + bias[4 * HID + ncol]));
      }
    }
    __syncthreads();

    // Phase 8: out_pre = (s_mid @ Ws2 + bs2) * pos  -> f32 LDS scratch.
    {
      v8f acc = {0.f, 0.f, 0.f, 0.f, 0.f, 0.f, 0.f, 0.f};
#pragma unroll
      for (int kt = 0; kt < 4; ++kt)
        acc = wmma_bf16(load_a_frag(midA, lrow, HID, kt * 32, half),
                        load_b_packed(wsm2, wave, 4, kt, lane), acc);
#pragma unroll
      for (int rI = 0; rI < 8; ++rI) {
        int m = rI + half * 8;
        scr[m * HID + ncol] = (acc[rI] + bias[5 * HID + ncol]) * posr[rI];
      }
    }
    __syncthreads();

    // Phase 9: attention gate per edge: sigmoid(out_pre . W_att + b_att).
    if (tid < 16) {
      float s = battv;
      for (int n = 0; n < HID; ++n) s += scr[tid * HID + n] * l_watt[n];
      attv[tid] = 1.0f / (1.0f + __expf(-s));
    }
    __syncthreads();

    // Phase 10: out = out_pre * att.
    for (int i = tid; i < 16 * HID; i += 256) {
      out0[(size_t)(e0 + (i >> 7)) * HID + (i & 127)] = scr[i] * attv[i >> 7];
    }
    __syncthreads();
  }
}

// ---------------------------------------------------------------------------
// Host launcher.
// Workspace layout (floats): sums[N*9] | cnt[N] | nvecs[N*9] | wbf(102400 u16)
// ---------------------------------------------------------------------------
extern "C" void kernel_launch(void* const* d_in, const int* in_sizes, int n_in,
                              void* d_out, int out_size, void* d_ws, size_t ws_size,
                              hipStream_t stream) {
  const float*     h     = (const float*)d_in[0];
  const float*     coord = (const float*)d_in[1];
  const long long* edges = (const long long*)d_in[2];
  const float* Wce1 = (const float*)d_in[3];
  const float* bce1 = (const float*)d_in[4];
  const float* Wce2 = (const float*)d_in[5];
  const float* bce2 = (const float*)d_in[6];
  const float* Wpe1 = (const float*)d_in[7];
  const float* bpe1 = (const float*)d_in[8];
  const float* Wpe2 = (const float*)d_in[9];
  const float* bpe2 = (const float*)d_in[10];
  const float* Ws1  = (const float*)d_in[11];
  const float* bs1  = (const float*)d_in[12];
  const float* Ws2  = (const float*)d_in[13];
  const float* bs2  = (const float*)d_in[14];
  const float* Watt = (const float*)d_in[15];
  const float* batt = (const float*)d_in[16];

  float* out      = (float*)d_out;
  float* out0     = out;
  float* out_chem = out + (size_t)NE * HID;
  float* out_pos  = out + (size_t)NE * HID * 2;
  float* out_cdn  = out + (size_t)NE * HID * 3;

  float* wsf   = (float*)d_ws;
  float* sums  = wsf;                 // N*9
  float* cnt   = wsf + NN * 9;        // N
  float* nvecs = wsf + NN * 10;       // N*9
  unsigned short* wbf = (unsigned short*)(wsf + NN * 19);  // 102400 bf16

  int prep_total = (NN * 10 > 102400) ? NN * 10 : 102400;
  k_prep<<<(prep_total + 255) / 256, 256, 0, stream>>>(Wce1, Wce2, Wpe1, Wpe2,
                                                       Ws1, Ws2, wsf, wbf);
  k_geom<<<(NE + 255) / 256, 256, 0, stream>>>(coord, edges, sums, cnt, out_cdn);
  k_node<<<(NN + 255) / 256, 256, 0, stream>>>(sums, cnt, nvecs);
  k_edge_mlp<<<512, 256, SMEM_BYTES, stream>>>(h, coord, edges, nvecs, wbf,
                                               bce1, bce2, bpe1, bpe2, bs1, bs2,
                                               Watt, batt, out0, out_chem, out_pos);
}
